// Attention_32736240730828
// MI455X (gfx1250) — matrix operand order repaired and checked by reading, not compiled, not run on hardware
//
#include <hip/hip_runtime.h>
#include <hip/hip_bf16.h>

typedef __attribute__((ext_vector_type(16))) _Float16 v16h;
typedef __attribute__((ext_vector_type(8)))  _Float16 h8;
typedef __attribute__((ext_vector_type(4)))  _Float16 h4;
typedef __attribute__((ext_vector_type(8)))  float    v8f;
typedef __attribute__((ext_vector_type(4)))  unsigned int u32x4;
typedef __attribute__((ext_vector_type(8)))  int      i32x8;
typedef __attribute__((ext_vector_type(4)))  int      i32x4;

#define LANE (threadIdx.x & 31u)

#if __has_builtin(__builtin_amdgcn_tensor_load_to_lds)
#define HAVE_TDM 1
#else
#define HAVE_TDM 0
#endif

__device__ __forceinline__ v8f wmma16(v16h a, v16h b, v8f c) {
  return __builtin_amdgcn_wmma_f32_16x16x32_f16(false, a, false, b, (short)0, c,
                                                false, false);
}

// ---- fragment loaders (wave32 layouts per CDNA5 ISA 7.12.2) ----------------
__device__ __forceinline__ v16h frag_a(const _Float16* p, int ld) {
  unsigned row = LANE & 15u, hf = LANE >> 4;
  const _Float16* r = p + row * ld + hf * 8u;
  v16h a;
#pragma unroll
  for (int j = 0; j < 8; ++j) a[j] = r[j];          // K = hf*8 .. +7
#pragma unroll
  for (int j = 0; j < 8; ++j) a[8 + j] = r[16 + j]; // K = 16+hf*8 .. +7
  return a;
}

// B tile stored N-major in LDS: p[n*ld + k] -> K = hf*8..+7 and 16+hf*8..+7 per lane
__device__ __forceinline__ v16h frag_b(const _Float16* p, int ld) {
  unsigned col = LANE & 15u, hf = LANE >> 4;
  const _Float16* r = p + col * ld + hf * 8u;
  v16h b;
#pragma unroll
  for (int i = 0; i < 8; ++i) b[i] = r[i];
#pragma unroll
  for (int i = 0; i < 8; ++i) b[8 + i] = r[16 + i];
  return b;
}

#if HAVE_TDM
// ---- TDM 2D tile load: global (row-major, f16) -> LDS with pitch 72 halfs --
// Pads 4 DWORDs after every 32 DWORDs (128B row) => LDS pitch 144B = 72 halfs.
// D# per CDNA5 ISA ch.8: group0 {count,lds,global,type=2}, group1 {dims/tiles}.
__device__ __forceinline__ void tdm_load_2d(unsigned lds_addr,
                                            const _Float16* gptr,
                                            unsigned tdim0, unsigned stride0,
                                            unsigned tdim1,
                                            unsigned tile0, unsigned tile1) {
  unsigned long long ga = (unsigned long long)(size_t)(const void*)gptr;
  u32x4 g0;
  g0[0] = 1u;                                         // count=1, user mode
  g0[1] = lds_addr;                                   // LDS byte address
  g0[2] = (unsigned)(ga & 0xFFFFFFFFu);               // global_addr[31:0]
  g0[3] = ((unsigned)(ga >> 32) & 0x01FFFFFFu) | 0x80000000u; // [56:32]|type=2
  i32x8 g1;
  // data_size=1 (2B) | pad_enable | pad_interval=4 (32 DW) | pad_amount=3 (4 DW)
  g1[0] = (int)((1u << 16) | (1u << 20) | (4u << 22) | (3u << 25));
  g1[1] = (int)(tdim0 << 16);                         // tensor_dim0[15:0]
  g1[2] = (int)((tdim0 >> 16) | (tdim1 << 16));       // dim0[31:16]|dim1[15:0]
  g1[3] = (int)((tdim1 >> 16) | (tile0 << 16));       // dim1[31:16]|tile_dim0
  g1[4] = (int)tile1;                                 // tile_dim1 (tile_dim2=0)
  g1[5] = (int)stride0;                               // tensor_dim0_stride[31:0]
  g1[6] = 0;
  g1[7] = 0;
  i32x4 z4 = {0, 0, 0, 0};
  i32x8 z8 = {0, 0, 0, 0, 0, 0, 0, 0};
  // 6-arg toolchain variant: (g0, g1, g2, g3, g4, cpol)
  __builtin_amdgcn_tensor_load_to_lds(g0, g1, z4, z4, z8, 0);
}
#endif

// ============================================================================
// Kernel 1: past (B,2,H,2048,64) f32 -> f16 K/V caches [B,H,4096,64] pos 0..2047
// ============================================================================
__global__ void att_past_cvt(const float* __restrict__ past,
                             _Float16* __restrict__ kbuf,
                             _Float16* __restrict__ vbuf) {
  size_t idx = ((size_t)blockIdx.x * blockDim.x + threadIdx.x) * 4;
  float4 f = *(const float4*)&past[idx];
  unsigned dd = (unsigned)(idx & 63u);
  unsigned p  = (unsigned)((idx >> 6) & 2047u);
  unsigned h  = (unsigned)((idx >> 17) & 15u);
  unsigned t  = (unsigned)((idx >> 21) & 1u);
  unsigned b  = (unsigned)(idx >> 22);
  size_t dst = (((size_t)b * 16 + h) * 4096 + p) * 64 + dd;
  h4 hv = { (_Float16)f.x, (_Float16)f.y, (_Float16)f.z, (_Float16)f.w };
  if (t == 0) *(h4*)&kbuf[dst] = hv;
  else        *(h4*)&vbuf[dst] = hv;
}

// ---- shared 8-WMMA step with all fragments preloaded -----------------------
#define LDT 72

__device__ __forceinline__ void gemm_step(const _Float16* As, const _Float16* Bt,
                                          int wm, int wn, int kk, v8f acc[2][4]) {
  v16h a0 = frag_a(&As[(wm * 32) * LDT + kk], LDT);
  v16h a1 = frag_a(&As[(wm * 32 + 16) * LDT + kk], LDT);
  v16h b0 = frag_b(&Bt[(wn * 64 + 0)  * LDT + kk], LDT);
  v16h b1 = frag_b(&Bt[(wn * 64 + 16) * LDT + kk], LDT);
  v16h b2 = frag_b(&Bt[(wn * 64 + 32) * LDT + kk], LDT);
  v16h b3 = frag_b(&Bt[(wn * 64 + 48) * LDT + kk], LDT);
  acc[0][0] = wmma16(a0, b0, acc[0][0]);
  acc[1][0] = wmma16(a1, b0, acc[1][0]);
  acc[0][1] = wmma16(a0, b1, acc[0][1]);
  acc[1][1] = wmma16(a1, b1, acc[1][1]);
  acc[0][2] = wmma16(a0, b2, acc[0][2]);
  acc[1][2] = wmma16(a1, b2, acc[1][2]);
  acc[0][3] = wmma16(a0, b3, acc[0][3]);
  acc[1][3] = wmma16(a1, b3, acc[1][3]);
}

// ============================================================================
// Kernel 2: QKV GEMM (4096x1024)@(1024x3072)+b1 ; scatter Q/K/V
// ============================================================================
__global__ void __launch_bounds__(256)
att_qkv_gemm(const float* __restrict__ x, const float* __restrict__ w1,
             const float* __restrict__ b1,
             _Float16* __restrict__ qbuf, _Float16* __restrict__ kbuf,
             _Float16* __restrict__ vbuf, float* __restrict__ present) {
  __shared__ _Float16 As[128 * LDT];  // [m][k]
  __shared__ _Float16 Bt[128 * LDT];  // [n][k]
  const int n0 = blockIdx.x * 128;
  const int m0 = blockIdx.y * 128;
  const int tid = threadIdx.x;
  const int w = tid >> 5, wm = w >> 1, wn = w & 1;

  v8f acc[2][4];
  v8f vz = {};
#pragma unroll
  for (int i = 0; i < 2; ++i)
#pragma unroll
    for (int j = 0; j < 4; ++j) acc[i][j] = vz;

  for (int k0 = 0; k0 < 1024; k0 += 64) {
    for (int i = tid; i < 128 * 16; i += 256) {       // A: f32->f16, h4 stores
      int r = i >> 4, c4 = (i & 15) * 4;
      float4 f = *(const float4*)&x[(size_t)(m0 + r) * 1024 + k0 + c4];
      h4 hv = { (_Float16)f.x, (_Float16)f.y, (_Float16)f.z, (_Float16)f.w };
      *(h4*)&As[r * LDT + c4] = hv;
    }
    for (int i = tid; i < 64 * 32; i += 256) {        // B: transposed scatter
      int r = i >> 5, c4 = (i & 31) * 4;
      float4 f = *(const float4*)&w1[(size_t)(k0 + r) * 3072 + n0 + c4];
      Bt[(c4 + 0) * LDT + r] = (_Float16)f.x;
      Bt[(c4 + 1) * LDT + r] = (_Float16)f.y;
      Bt[(c4 + 2) * LDT + r] = (_Float16)f.z;
      Bt[(c4 + 3) * LDT + r] = (_Float16)f.w;
    }
    __syncthreads();
    gemm_step(As, Bt, wm, wn, 0, acc);
    gemm_step(As, Bt, wm, wn, 32, acc);
    __syncthreads();
  }

  const unsigned colL = LANE & 15u, rh = LANE >> 4;
#pragma unroll
  for (int mt = 0; mt < 2; ++mt)
#pragma unroll
    for (int nt = 0; nt < 4; ++nt)
#pragma unroll
      for (int r = 0; r < 8; ++r) {
        int grow = m0 + wm * 32 + mt * 16 + (int)rh * 8 + r;
        int gcol = n0 + wn * 64 + nt * 16 + (int)colL;
        float val = acc[mt][nt][r] + b1[gcol];
        int bb = grow >> 11, s = grow & 2047;
        int which = gcol >> 10, d = gcol & 1023;
        int h = d >> 6, dd = d & 63;
        if (which == 0) {
          qbuf[(((size_t)bb * 16 + h) * 2048 + s) * 64 + dd] = (_Float16)val;
        } else {
          present[((((size_t)bb * 2 + (which - 1)) * 16 + h) * 2048 + s) * 64 + dd] = val;
          _Float16* kv = (which == 1) ? kbuf : vbuf;
          kv[(((size_t)bb * 16 + h) * 4096 + 2048 + (size_t)s) * 64 + dd] = (_Float16)val;
        }
      }
}

// ============================================================================
// Kernel 3: flash attention per (b,h,128-row q tile); 64-key blocks, ns=4096
// Q and K tiles arrive via the Tensor Data Mover (padded to 72-half pitch).
// ============================================================================
#define LDH 72

__global__ void __launch_bounds__(256)
att_flash(const _Float16* __restrict__ qbuf, const _Float16* __restrict__ kbuf,
          const _Float16* __restrict__ vbuf, _Float16* __restrict__ abuf) {
  __shared__ _Float16 Qs[128 * LDH];     // [q][dh]
  __shared__ _Float16 Ks[64 * LDH];      // [key][dh]
  __shared__ _Float16 Vt[64 * LDH];      // [dh][key]
  __shared__ _Float16 Ps[8 * 16 * LDH];  // per-wave P strip

  const int qt = blockIdx.x, h = blockIdx.y, b = blockIdx.z;
  const int q0 = qt * 128;
  const int tid = threadIdx.x;
  const int w = tid >> 5;
  const unsigned colL = LANE & 15u, rh = LANE >> 4;

  const _Float16* Qg = qbuf + (((size_t)b * 16 + h) * 2048 + q0) * 64;
  const _Float16* Kg = kbuf + ((size_t)b * 16 + h) * 4096 * 64;
  const _Float16* Vg = vbuf + ((size_t)b * 16 + h) * 4096 * 64;

#if HAVE_TDM
  if (w == 0)
    tdm_load_2d((unsigned)(size_t)(void*)Qs, Qg, 64, 64, 2048, 64, 128);
#else
  for (int i = tid; i < 128 * 8; i += 256) {
    int r = i >> 3, c8 = (i & 7) * 8;
    *(h8*)&Qs[r * LDH + c8] = *(const h8*)&Qg[(size_t)r * 64 + c8];
  }
#endif

  float mrow[8], lrow[8];
  v8f o[4];
  v8f vz = {};
#pragma unroll
  for (int r = 0; r < 8; ++r) { mrow[r] = -1e30f; lrow[r] = 0.f; }
#pragma unroll
  for (int dt = 0; dt < 4; ++dt) o[dt] = vz;

  int nkb = (q0 + 127 + 2048) / 64 + 1;
  if (nkb > 64) nkb = 64;
  _Float16* Pw = &Ps[w * 16 * LDH];
  const int qr0 = w * 16;

  for (int kb = 0; kb < nkb; ++kb) {
#if HAVE_TDM
    if (w == 0)
      tdm_load_2d((unsigned)(size_t)(void*)Ks, Kg + (size_t)kb * 64 * 64,
                  64, 64, 4096, 64, 64);
#else
    for (int i = tid; i < 64 * 8; i += 256) {
      int r = i >> 3, c8 = (i & 7) * 8;
      *(h8*)&Ks[r * LDH + c8] = *(const h8*)&Kg[((size_t)kb * 64 + r) * 64 + c8];
    }
#endif
    for (int i = tid; i < 64 * 8; i += 256) {          // V transposed into Vt
      int r = i >> 3, c8 = (i & 7) * 8;
      h8 hv = *(const h8*)&Vg[((size_t)kb * 64 + r) * 64 + c8];
#pragma unroll
      for (int j = 0; j < 8; ++j) Vt[(c8 + j) * LDH + r] = hv[j];
    }
    if (kb + 1 < nkb && tid < 64) {
      __builtin_prefetch(&Kg[((size_t)(kb + 1) * 64 + tid) * 64], 0, 0);
      __builtin_prefetch(&Vg[((size_t)(kb + 1) * 64 + tid) * 64], 0, 0);
    }
#if HAVE_TDM
    if (w == 0) __builtin_amdgcn_s_wait_tensorcnt(0);  // Q (first iter) + K
#endif
    __syncthreads();

    // S = Q(16x64) x K^T(64x64): preload all fragments per K-step
    v8f sc[4];
#pragma unroll
    for (int nt = 0; nt < 4; ++nt) sc[nt] = vz;
#pragma unroll
    for (int kk = 0; kk < 64; kk += 32) {
      v16h aq = frag_a(&Qs[qr0 * LDH + kk], LDH);
      v16h k0f = frag_b(&Ks[0  * LDH + kk], LDH);
      v16h k1f = frag_b(&Ks[16 * LDH + kk], LDH);
      v16h k2f = frag_b(&Ks[32 * LDH + kk], LDH);
      v16h k3f = frag_b(&Ks[48 * LDH + kk], LDH);
      sc[0] = wmma16(aq, k0f, sc[0]);
      sc[1] = wmma16(aq, k1f, sc[1]);
      sc[2] = wmma16(aq, k2f, sc[2]);
      sc[3] = wmma16(aq, k3f, sc[3]);
    }

    // scale + causal mask + running row max
    float mx[8];
#pragma unroll
    for (int r = 0; r < 8; ++r) mx[r] = -1e30f;
#pragma unroll
    for (int nt = 0; nt < 4; ++nt) {
      int j = kb * 64 + nt * 16 + (int)colL;
#pragma unroll
      for (int r = 0; r < 8; ++r) {
        int qrow = q0 + qr0 + (int)rh * 8 + r;
        float v = sc[nt][r] * 0.125f;
        v = (j <= qrow + 2048) ? v : -1e30f;
        sc[nt][r] = v;
        mx[r] = fmaxf(mx[r], v);
      }
    }
#pragma unroll
    for (int m = 1; m < 16; m <<= 1)
#pragma unroll
      for (int r = 0; r < 8; ++r)
        mx[r] = fmaxf(mx[r], __shfl_xor(mx[r], m, 32));

    float alpha[8], rs[8];
#pragma unroll
    for (int r = 0; r < 8; ++r) {
      float mn = fmaxf(mrow[r], mx[r]);
      alpha[r] = __expf(mrow[r] - mn);
      mrow[r] = mn;
      rs[r] = 0.f;
    }
#pragma unroll
    for (int nt = 0; nt < 4; ++nt)
#pragma unroll
      for (int r = 0; r < 8; ++r) {
        float p = __expf(sc[nt][r] - mrow[r]);
        sc[nt][r] = p;
        rs[r] += p;
      }
#pragma unroll
    for (int m = 1; m < 16; m <<= 1)
#pragma unroll
      for (int r = 0; r < 8; ++r) rs[r] += __shfl_xor(rs[r], m, 32);
#pragma unroll
    for (int r = 0; r < 8; ++r) lrow[r] = lrow[r] * alpha[r] + rs[r];
#pragma unroll
    for (int dt = 0; dt < 4; ++dt)
#pragma unroll
      for (int r = 0; r < 8; ++r) o[dt][r] *= alpha[r];

    // P -> per-wave LDS strip (D-layout -> A-layout)
#pragma unroll
    for (int nt = 0; nt < 4; ++nt)
#pragma unroll
      for (int r = 0; r < 8; ++r)
        Pw[(rh * 8u + r) * LDH + nt * 16 + colL] = (_Float16)sc[nt][r];

    // O += P(16x64) x V(64x64)
#pragma unroll
    for (int kk = 0; kk < 64; kk += 32) {
      v16h ap = frag_a(&Pw[kk], LDH);
      v16h v0f = frag_b(&Vt[0  * LDH + kk], LDH);
      v16h v1f = frag_b(&Vt[16 * LDH + kk], LDH);
      v16h v2f = frag_b(&Vt[32 * LDH + kk], LDH);
      v16h v3f = frag_b(&Vt[48 * LDH + kk], LDH);
      o[0] = wmma16(ap, v0f, o[0]);
      o[1] = wmma16(ap, v1f, o[1]);
      o[2] = wmma16(ap, v2f, o[2]);
      o[3] = wmma16(ap, v3f, o[3]);
    }
    __syncthreads();
  }

#pragma unroll
  for (int r = 0; r < 8; ++r) lrow[r] = 1.f / lrow[r];
#pragma unroll
  for (int dt = 0; dt < 4; ++dt)
#pragma unroll
    for (int r = 0; r < 8; ++r) {
      int s = q0 + qr0 + (int)rh * 8 + r;
      int dd = dt * 16 + (int)colL;
      abuf[((size_t)b * 2048 + s) * 1024 + h * 64 + dd] =
          (_Float16)(o[dt][r] * lrow[r]);
    }
}

// ============================================================================
// Kernel 4: out = a @ w2 + b2  (4096x1024)@(1024x1024), f32 result
// A tile (already f16) arrives via TDM; B tile transposed manually.
// ============================================================================
__global__ void __launch_bounds__(256)
att_out_gemm(const _Float16* __restrict__ abuf, const float* __restrict__ w2,
             const float* __restrict__ b2, float* __restrict__ out) {
  __shared__ _Float16 As[128 * LDT];
  __shared__ _Float16 Bt[128 * LDT];
  const int n0 = blockIdx.x * 128;
  const int m0 = blockIdx.y * 128;
  const int tid = threadIdx.x;
  const int w = tid >> 5, wm = w >> 1, wn = w & 1;

  v8f acc[2][4];
  v8f vz = {};
#pragma unroll
  for (int i = 0; i < 2; ++i)
#pragma unroll
    for (int j = 0; j < 4; ++j) acc[i][j] = vz;

  for (int k0 = 0; k0 < 1024; k0 += 64) {
#if HAVE_TDM
    if (w == 0)
      tdm_load_2d((unsigned)(size_t)(void*)As, &abuf[(size_t)m0 * 1024 + k0],
                  1024, 1024, 4096, 64, 128);
#else
    for (int i = tid; i < 128 * 8; i += 256) {
      int r = i >> 3, c8 = (i & 7) * 8;
      *(h8*)&As[r * LDT + c8] = *(const h8*)&abuf[(size_t)(m0 + r) * 1024 + k0 + c8];
    }
#endif
    for (int i = tid; i < 64 * 32; i += 256) {
      int r = i >> 5, c4 = (i & 31) * 4;
      float4 f = *(const float4*)&w2[(size_t)(k0 + r) * 1024 + n0 + c4];
      Bt[(c4 + 0) * LDT + r] = (_Float16)f.x;
      Bt[(c4 + 1) * LDT + r] = (_Float16)f.y;
      Bt[(c4 + 2) * LDT + r] = (_Float16)f.z;
      Bt[(c4 + 3) * LDT + r] = (_Float16)f.w;
    }
#if HAVE_TDM
    if (w == 0) __builtin_amdgcn_s_wait_tensorcnt(0);
#endif
    __syncthreads();
    gemm_step(As, Bt, wm, wn, 0, acc);
    gemm_step(As, Bt, wm, wn, 32, acc);
    __syncthreads();
  }

  const unsigned colL = LANE & 15u, rh = LANE >> 4;
#pragma unroll
  for (int mt = 0; mt < 2; ++mt)
#pragma unroll
    for (int nt = 0; nt < 4; ++nt)
#pragma unroll
      for (int r = 0; r < 8; ++r) {
        int grow = m0 + wm * 32 + mt * 16 + (int)rh * 8 + r;
        int gcol = n0 + wn * 64 + nt * 16 + (int)colL;
        out[(size_t)grow * 1024 + gcol] = acc[mt][nt][r] + b2[gcol];
      }
}

// ============================================================================
extern "C" void kernel_launch(void* const* d_in, const int* in_sizes, int n_in,
                              void* d_out, int out_size, void* d_ws, size_t ws_size,
                              hipStream_t stream) {
  const float* x    = (const float*)d_in[0];
  const float* past = (const float*)d_in[1];
  const float* w1   = (const float*)d_in[2];
  const float* b1   = (const float*)d_in[3];
  const float* w2   = (const float*)d_in[4];
  const float* b2   = (const float*)d_in[5];

  float* out     = (float*)d_out;
  float* present = out + (size_t)2 * 2048 * 1024;

  _Float16* qbuf = (_Float16*)d_ws;                        // [B,H,2048,64]
  _Float16* kbuf = qbuf + (size_t)2 * 16 * 2048 * 64;      // [B,H,4096,64]
  _Float16* vbuf = kbuf + (size_t)2 * 16 * 4096 * 64;      // [B,H,4096,64]
  _Float16* abuf = vbuf + (size_t)2 * 16 * 4096 * 64;      // [B,S,1024]

  att_past_cvt<<<dim3(16777216 / 4 / 256), dim3(256), 0, stream>>>(past, kbuf, vbuf);
  att_qkv_gemm<<<dim3(24, 32), dim3(256), 0, stream>>>(x, w1, b1, qbuf, kbuf,
                                                       vbuf, present);
  att_flash<<<dim3(16, 16, 2), dim3(256), 0, stream>>>(qbuf, kbuf, vbuf, abuf);
  att_out_gemm<<<dim3(8, 32), dim3(256), 0, stream>>>(abuf, w2, b2, out);
}
